// FactorLayer_28827820491136
// MI455X (gfx1250) — compile-verified, matrix operand order checked by reading, not run on hardware
//
#include <hip/hip_runtime.h>
#include <hip/hip_bf16.h>
#include <math.h>
#include <stdint.h>

typedef __bf16 bf16_t;
typedef __attribute__((ext_vector_type(16))) __bf16 v16bf;
typedef __attribute__((ext_vector_type(8)))  float  v8f;
typedef __attribute__((ext_vector_type(4)))  int    v4i;

#define BATCH  4096
#define FDIM   4096
#define NDLV   32
#define KTILES (FDIM / 32)      // 128 k-tiles of 32
#define LEPS   0.001f

// ---- workspace layout (float offsets) ----
#define OFF_COLSUM 0            // [4096]
#define OFF_COLSQ  4096         // [4096]
#define OFF_SCALE  8192         // [4096]
#define OFF_SHIFT  12288        // [4096]
#define OFF_BIAS   16384        // [64]
#define OFF_GRAM   16448        // [64*64]
#define OFF_SVEC   20544        // [64]
#define OFF_P      20608        // [32*32]
#define OFF_Q      21632        // [32]
#define OFF_Y      21664        // [4096*64]
#define OFF_F32_END (21664 + 4096*64)
#define NZCLR      20608        // zero colsum..svec
#define BF_BASE_BYTES ((size_t)OFF_F32_END * 4)
#define NFRAG      (KTILES * 4 * 32 * 16)   // 262144 bf16 per (hi|lo)

// ---------------------------------------------------------------- feature probes
#ifdef __has_builtin
#define HASB(x) __has_builtin(x)
#else
#define HASB(x) 0
#endif

#if HASB(__builtin_amdgcn_global_load_async_to_lds_b128)
#define HAVE_ASYNC_LDS 1
#else
#define HAVE_ASYNC_LDS 0
#warning "gfx1250 async-to-LDS builtin unavailable in this pass; using sync LDS staging"
#endif

#if HAVE_ASYNC_LDS
// GLOBAL_LOAD_ASYNC_TO_LDS_B128: per-lane 16B memory -> LDS, tracked by ASYNCcnt.
// Builtin signature (from probe diagnostics): (v4i AS1*, v4i AS3*, imm int, imm int).
// AS(1) from flat global is value-identical; AS(3) = low 32 bits of flat LDS addr
__device__ __forceinline__ void async_b128(const float* g, float* l) {
  __builtin_amdgcn_global_load_async_to_lds_b128(
      (__attribute__((address_space(1))) v4i*)(uintptr_t)g,
      (__attribute__((address_space(3))) v4i*)(unsigned)(uintptr_t)l,
      0, 0);
}
#endif

template <int N>
__device__ __forceinline__ void wait_asynccnt() {
#if HASB(__builtin_amdgcn_s_wait_asynccnt)
  __builtin_amdgcn_s_wait_asynccnt(N);
#else
  asm volatile("s_wait_asynccnt %0" ::"i"(N) : "memory");
#endif
}

// ---------------------------------------------------------------- utilities
__device__ __forceinline__ float wred32(float v) {
#pragma unroll
  for (int m = 16; m >= 1; m >>= 1) v += __shfl_xor(v, m, 32);
  return v;
}

// ---------------------------------------------------------------- K0: zero
__global__ void zero_kernel(float* __restrict__ p, int n) {
  int i = blockIdx.x * 256 + threadIdx.x;
  if (i < n) p[i] = 0.f;
}

// ------------------------------------------------- K1: per-feature batch stats
__global__ void stats_kernel(const float* __restrict__ in,
                             float* __restrict__ cs, float* __restrict__ cq) {
  int col = blockIdx.x * 256 + threadIdx.x;          // feature
  int r0  = blockIdx.y * 128;                        // row chunk
  float s = 0.f, s2 = 0.f;
  for (int r = 0; r < 128; ++r) {
    float v = in[(size_t)(r0 + r) * FDIM + col];
    s += v; s2 += v * v;
  }
  atomicAdd(&cs[col], s);
  atomicAdd(&cq[col], s2);
}

// ------------------------------------------------- K2: scale/shift per feature
__global__ void finalize_stats(const float* __restrict__ cs, const float* __restrict__ cq,
                               const float* __restrict__ gamma, const float* __restrict__ beta,
                               float* __restrict__ scale, float* __restrict__ shift) {
  int f = blockIdx.x * 256 + threadIdx.x;
  float mean = cs[f] * (1.f / (float)BATCH);
  float var  = cq[f] * (1.f / (float)BATCH) - mean * mean;
  float sc   = rsqrtf(var + LEPS) * gamma[f];
  scale[f] = sc;
  shift[f] = beta[f] - mean * sc;
}

// ------------------------------------------------- K3: build WMMA B fragments
// W' = scale ⊙ [W_static | W_train], split into bf16 hi+lo, stored directly in
// the wave32 v_wmma_f32_16x16x32_bf16 B-operand layout:
//   frag index = ((kt*4 + nt)*32 + lane)*16 + elem,
//   lane = n + 16*((kk>>3)&1), elem = ((kk&16)?8:0) + (kk&7)
// Also accumulates bias_j = sum_f shift_f * W[f,j].
__global__ void prep_w_kernel(const float* __restrict__ Ws, const float* __restrict__ Wt,
                              const float* __restrict__ scale, const float* __restrict__ shift,
                              bf16_t* __restrict__ bhi, bf16_t* __restrict__ blo,
                              float* __restrict__ bias) {
  __shared__ float bpart[64];
  int t = threadIdx.x;
  if (t < 64) bpart[t] = 0.f;
  __syncthreads();

  int f = blockIdx.x * 256 + t;
  float sc = scale[f], sh = shift[f];
  int kt   = f >> 5;
  int kk   = f & 31;
  int half = (kk & 8) ? 1 : 0;
  int elem = ((kk & 16) ? 8 : 0) + (kk & 7);
#pragma unroll 4
  for (int j = 0; j < 64; ++j) {
    float w  = (j < 32) ? Ws[(size_t)f * 32 + j] : Wt[(size_t)f * 32 + (j - 32)];
    float wp = sc * w;
    bf16_t h = (bf16_t)wp;
    bf16_t l = (bf16_t)(wp - (float)h);
    int nt = j >> 4, n = j & 15;
    size_t idx = (((size_t)kt * 4 + nt) * 32 + (n + 16 * half)) * 16 + elem;
    bhi[idx] = h;
    blo[idx] = l;
    atomicAdd(&bpart[j], sh * w);
  }
  __syncthreads();
  if (t < 64) atomicAdd(&bias[t], bpart[t]);
}

// ------------------------------------------------- K4: WMMA GEMM  Y = in @ W' + bias
// 128 threads = 4 waves; block owns a 16-row M tile and all 64 columns
// (one 16-col tile per wave). K loop: 128 iterations of K=32.
// fp32 emulated via bf16 split: c += Ahi*Bhi + Ahi*Blo + Alo*Bhi.
// A tile is staged memory->LDS with double-buffered GLOBAL_LOAD_ASYNC_TO_LDS_B128
// (ASYNCcnt) when available.
__global__ __launch_bounds__(128) void gemm_wmma_kernel(
    const float* __restrict__ in,
    const bf16_t* __restrict__ bhi, const bf16_t* __restrict__ blo,
    const float* __restrict__ bias, float* __restrict__ Y) {
  __shared__ float At[2][16 * 36];              // 2 x (16x32 tile, stride 36)
  int t    = threadIdx.x;
  int lane = t & 31;
  int wv   = t >> 5;                            // n-tile 0..3
  int m0   = blockIdx.x * 16;

  // cooperative A-tile staging indices (16B per thread)
  int arow = t >> 3;                            // 0..15
  int acol = (t & 7) * 4;                       // 0..28
  const float* gA = in + (size_t)(m0 + arow) * FDIM + acol;
  float* l0 = &At[0][arow * 36 + acol];
  float* l1 = &At[1][arow * 36 + acol];

  // per-lane A-fragment gather indices (WMMA 16-bit A layout)
  int m  = lane & 15;
  int kb = (lane >> 4) * 8;                     // 0 or 8

  const bf16_t* fbh = bhi + ((size_t)wv * 32 + lane) * 16;
  const bf16_t* fbl = blo + ((size_t)wv * 32 + lane) * 16;

  v8f c = {};
#if HAVE_ASYNC_LDS
  async_b128(gA, l0);                           // prologue: tile 0 -> buf 0
#endif
  for (int kt = 0; kt < KTILES; ++kt) {
    const float* cur = At[kt & 1];
#if HAVE_ASYNC_LDS
    if (kt + 1 < KTILES) {
      async_b128(gA + (size_t)(kt + 1) * 32, ((kt + 1) & 1) ? l1 : l0);
      wait_asynccnt<1>();                       // oldest (tile kt) complete
    } else {
      wait_asynccnt<0>();
    }
    __syncthreads();                            // tile kt visible to all waves
#else
    float4 av = *(const float4*)(gA + (size_t)kt * 32);
    __syncthreads();
    *(float4*)((kt & 1) ? l1 : l0) = av;
    __syncthreads();
#endif

    const float* fr0 = cur + m * 36 + kb;       // K = kb..kb+7
    const float* fr1 = fr0 + 16;                // K = kb+16..kb+23
    v16bf ahi, alo;
#pragma unroll
    for (int e = 0; e < 8; ++e) {
      float x0 = fr0[e];
      float x1 = fr1[e];
      bf16_t h0 = (bf16_t)x0, h1 = (bf16_t)x1;
      ahi[e]     = h0;  ahi[e + 8] = h1;
      alo[e]     = (bf16_t)(x0 - (float)h0);
      alo[e + 8] = (bf16_t)(x1 - (float)h1);
    }
    v16bf wh = *(const v16bf*)(fbh + (size_t)kt * 4 * 32 * 16);
    v16bf wl = *(const v16bf*)(fbl + (size_t)kt * 4 * 32 * 16);

    c = __builtin_amdgcn_wmma_f32_16x16x32_bf16(false, ahi, false, wh, (short)0, c, false, false);
    c = __builtin_amdgcn_wmma_f32_16x16x32_bf16(false, ahi, false, wl, (short)0, c, false, false);
    c = __builtin_amdgcn_wmma_f32_16x16x32_bf16(false, alo, false, wh, (short)0, c, false, false);

    __syncthreads();                            // reads of buf done before reuse
  }

  // C/D layout: VGPR r -> M = r + 8*(lane>=16), N = lane&15
  int n  = lane & 15;
  int mh = (lane >> 4) * 8;
  float bv = bias[wv * 16 + n];
#pragma unroll
  for (int r = 0; r < 8; ++r)
    Y[(size_t)(m0 + mh + r) * 64 + wv * 16 + n] = c[r] + bv;
}

// ------------------------------------------------- K5: Gram matrix + colsums of Y
__global__ void gram_kernel(const float* __restrict__ Y,
                            float* __restrict__ Mm, float* __restrict__ sv) {
  int t = threadIdx.x;
  int p = blockIdx.x * 1024 + t;                // (i,j) pair
  int i = p >> 6, j = p & 63;
  int r0 = blockIdx.y * 128;
  float acc = 0.f;
  for (int r = 0; r < 128; ++r) {
    const float* row = Y + (size_t)(r0 + r) * 64;
    acc += row[i] * row[j];
  }
  atomicAdd(&Mm[p], acc);
  if (blockIdx.x == 0 && t < 64) {
    float a = 0.f;
    for (int r = 0; r < 128; ++r) a += Y[(size_t)(r0 + r) * 64 + t];
    atomicAdd(&sv[t], a);
  }
}

// ------------------------------------------------- K6: coefficient recurrence
// Single wave32. Runs the 32-step orthogonalization recurrence entirely in
// Gram space: S_i ~ (a_i over Ys columns, const c_i). Outputs P[32x32], q[32]
// with out = Yt + Ys@P + 1 qᵀ.
__global__ __launch_bounds__(32) void coef_kernel(
    const float* __restrict__ Mm, const float* __restrict__ sv,
    float* __restrict__ P, float* __restrict__ q) {
  __shared__ float aP[32 * 32];   // standardized coefficient vectors a'_j
  __shared__ float cP[32];        // standardized constants c'_j
  __shared__ float aL[32];
  int k = threadIdx.x;
  const float Bf = (float)BATCH;
  const float rB = 1.f / Bf;

  for (int i = 0; i < NDLV; ++i) {
    float accA = 0.f, accC = 0.f, pacc = 0.f, qacc = 0.f;
    for (int j = 0; j < i; ++j) {
      float apj = aP[j * 32 + k];
      float cpj = cP[j];
      float ps = wred32(apj * Mm[k * 64 + i])      + cpj * sv[i];        // d_j · Ys e_i
      float pt = wred32(apj * Mm[k * 64 + 32 + i]) + cpj * sv[32 + i];   // d_j · Yt e_i
      float csf = ps * rB, ctf = pt * rB;
      accA -= csf * apj;  accC -= csf * cpj;
      pacc -= ctf * apj;  qacc -= ctf * cpj;
    }
    float ai = ((k == i) ? 1.f : 0.f) + accA;
    float ci = accC;
    P[k * 32 + i] = pacc;
    if (k == 0) q[i] = qacc;

    __syncthreads();
    aL[k] = ai;
    __syncthreads();
    float dg = wred32(ai * sv[k]);                   // a_i · g1
    float v = 0.f;
#pragma unroll 4
    for (int m2 = 0; m2 < 32; ++m2) v += Mm[k * 64 + m2] * aL[m2];   // (G a_i)_k
    float aGa  = wred32(ai * v);
    float mean = (dg + ci * Bf) * rB;
    float ssq  = aGa + 2.f * ci * dg + ci * ci * Bf;
    float var  = ssq * rB - mean * mean;
    float inv  = 1.f / (sqrtf(var) + LEPS);
    __syncthreads();
    aP[i * 32 + k] = ai * inv;
    if (k == 0) cP[i] = (ci - mean) * inv;
    __syncthreads();
  }
}

// ------------------------------------------------- K7: out = Yt + Ys@P + q
__global__ void out_kernel(const float* __restrict__ Y, const float* __restrict__ P,
                           const float* __restrict__ q, float* __restrict__ out) {
  __shared__ float Pl[32 * 32];
  __shared__ float ql[32];
  int t = threadIdx.x;
  for (int idx = t; idx < 1024; idx += 256) Pl[idx] = P[idx];
  if (t < 32) ql[t] = q[t];
  __syncthreads();
  int r = t >> 5, i = t & 31;
  int b = blockIdx.x * 8 + r;
  const float* yrow = Y + (size_t)b * 64;
  float acc = ql[i];
#pragma unroll 4
  for (int k2 = 0; k2 < 32; ++k2) acc += yrow[k2] * Pl[k2 * 32 + i];
  out[(size_t)b * 32 + i] = yrow[32 + i] + acc;
}

// ---------------------------------------------------------------- launcher
extern "C" void kernel_launch(void* const* d_in, const int* in_sizes, int n_in,
                              void* d_out, int out_size, void* d_ws, size_t ws_size,
                              hipStream_t stream) {
  const float* in    = (const float*)d_in[0];
  const float* gamma = (const float*)d_in[1];
  const float* beta  = (const float*)d_in[2];
  const float* Ws    = (const float*)d_in[3];
  const float* Wt    = (const float*)d_in[4];

  float* ws      = (float*)d_ws;
  float* colsum  = ws + OFF_COLSUM;
  float* colsq   = ws + OFF_COLSQ;
  float* scale   = ws + OFF_SCALE;
  float* shift   = ws + OFF_SHIFT;
  float* bias    = ws + OFF_BIAS;
  float* Mm      = ws + OFF_GRAM;
  float* sv      = ws + OFF_SVEC;
  float* P       = ws + OFF_P;
  float* q       = ws + OFF_Q;
  float* Y       = ws + OFF_Y;
  bf16_t* bhi    = (bf16_t*)((char*)d_ws + BF_BASE_BYTES);
  bf16_t* blo    = bhi + NFRAG;

  zero_kernel<<<(NZCLR + 255) / 256, 256, 0, stream>>>(ws, NZCLR);
  stats_kernel<<<dim3(16, 32), 256, 0, stream>>>(in, colsum, colsq);
  finalize_stats<<<16, 256, 0, stream>>>(colsum, colsq, gamma, beta, scale, shift);
  prep_w_kernel<<<16, 256, 0, stream>>>(Ws, Wt, scale, shift, bhi, blo, bias);
  gemm_wmma_kernel<<<BATCH / 16, 128, 0, stream>>>(in, bhi, blo, bias, Y);
  gram_kernel<<<dim3(4, 32), 1024, 0, stream>>>(Y, Mm, sv);
  coef_kernel<<<1, 32, 0, stream>>>(Mm, sv, P, q);
  out_kernel<<<BATCH / 8, 256, 0, stream>>>(Y, P, q, (float*)d_out);
}
